// Quantizer_33887291965458
// MI455X (gfx1250) — matrix operand order repaired and checked by reading, compiled, not run on hardware
//
#include <hip/hip_runtime.h>

typedef __bf16 bf16;
typedef __attribute__((ext_vector_type(16))) __bf16 v16bf;
typedef __attribute__((ext_vector_type(8)))  __bf16 v8bf;
typedef __attribute__((ext_vector_type(8)))  float  v8f;
typedef __attribute__((ext_vector_type(4)))  int    v4i;

union BFrag { v16bf v; v8bf h[2]; };

#define NUM_EMB 8192
#define EMB_DIM 256
#define NTOK    16384          // 16 * 32 * 32
#define NELEM   4194304        // NTOK * EMB_DIM
#define LDS_ROW 264            // 256 + 8 pad -> conflict-free ds_load_b128

// ---- gfx1250 async memory->LDS path (ASYNCcnt), with compile-safe fallback.
// Builtin signature (from hipcc diagnostic): (AS1 v4i*, AS3 v4i*, Iint, Iint)
#if __has_builtin(__builtin_amdgcn_global_load_async_to_lds_b128) && \
    __has_builtin(__builtin_amdgcn_s_wait_asynccnt)
#define USE_ASYNC_LDS 1
#define GASYNC_B128(gsrc, ldst)                                            \
    __builtin_amdgcn_global_load_async_to_lds_b128(                        \
        (__attribute__((address_space(1))) v4i*)(v4i*)(void*)(gsrc),       \
        (__attribute__((address_space(3))) v4i*)(v4i*)(void*)(ldst), 0, 0)
#else
#define USE_ASYNC_LDS 0
#endif

// ---------------------------------------------------------------------------
// Pass 1: z [B,D,H,W] f32 -> token-major bf16 hi/lo [N,D]
// ---------------------------------------------------------------------------
__global__ __launch_bounds__(256) void pack_z_kernel(const float* __restrict__ z,
                                                     bf16* __restrict__ zhi,
                                                     bf16* __restrict__ zlo) {
    const size_t i = (size_t)blockIdx.x * 256 + threadIdx.x;   // linear in z order
    const float x = z[i];
    const bf16 h = (bf16)x;
    const bf16 l = (bf16)(x - (float)h);
    const size_t b  = i >> 18;          // / (256*1024)
    const size_t d  = (i >> 10) & 255;
    const size_t hw = i & 1023;
    const size_t t  = b * 1024 + hw;
    zhi[t * EMB_DIM + d] = h;
    zlo[t * EMB_DIM + d] = l;
}

// ---------------------------------------------------------------------------
// Pass 2: codebook f32 -> bf16 hi/lo (same row-major layout)
// ---------------------------------------------------------------------------
__global__ __launch_bounds__(256) void pack_w_kernel(const float* __restrict__ w,
                                                     bf16* __restrict__ whi,
                                                     bf16* __restrict__ wlo) {
    const size_t i = (size_t)blockIdx.x * 256 + threadIdx.x;
    const float x = w[i];
    const bf16 h = (bf16)x;
    const bf16 l = (bf16)(x - (float)h);
    whi[i] = h;
    wlo[i] = l;
}

// ---------------------------------------------------------------------------
// Pass 3: ||w||^2 per code (exact fp32), one wave32 per code row
// ---------------------------------------------------------------------------
__global__ __launch_bounds__(256) void wsq_kernel(const float* __restrict__ w,
                                                  float* __restrict__ wsq) {
    const int k    = blockIdx.x * 8 + (threadIdx.x >> 5);
    const int lane = threadIdx.x & 31;
    float s = 0.0f;
#pragma unroll
    for (int j = 0; j < 8; ++j) {
        const float v = w[(size_t)k * EMB_DIM + lane + j * 32];
        s += v * v;
    }
#pragma unroll
    for (int m = 16; m >= 1; m >>= 1) s += __shfl_xor(s, m, 32);
    if (lane == 0) wsq[k] = s;
}

// ---------------------------------------------------------------------------
// Pass 4: WMMA distance + argmin.
//   block = 256 threads = 8 waves; wave -> 16-token tile; block -> 128 tokens.
//   Double-buffered 16-code LDS panel, staged via global_load_async_to_lds
//   (tile ct+1 copies overlap WMMA on tile ct). score = ||w||^2 - 2*(z.w)
//   via 3-term bf16 hi/lo split WMMA (f32 accumulate).
// ---------------------------------------------------------------------------
__global__ __launch_bounds__(256) void vq_argmin_kernel(const bf16* __restrict__ zhi,
                                                        const bf16* __restrict__ zlo,
                                                        const bf16* __restrict__ whi,
                                                        const bf16* __restrict__ wlo,
                                                        const float* __restrict__ wsq,
                                                        int* __restrict__ ids) {
    __shared__ bf16 sWhi[2][16 * LDS_ROW];
    __shared__ bf16 sWlo[2][16 * LDS_ROW];

    const int tid  = threadIdx.x;
    const int wave = tid >> 5;
    const int lane = tid & 31;
    const int n    = lane & 15;     // col (code within tile) for B/C; row for A
    const int hh   = lane >> 4;     // half-wave select

    const int tokBase = blockIdx.x * 128 + wave * 16;

    // ---- A fragments (persistent in VGPRs): 16x32 bf16 layout per k-block.
    // lanes 0-15: K = {0..7, 16..23}; lanes 16-31: K = {8..15, 24..31}.
    BFrag Ahi[8], Alo[8];
    {
        const bf16* zrh = zhi + (size_t)(tokBase + n) * EMB_DIM;
        const bf16* zrl = zlo + (size_t)(tokBase + n) * EMB_DIM;
#pragma unroll
        for (int kb = 0; kb < 8; ++kb) {
            const int d0 = kb * 32 + hh * 8;
            Ahi[kb].h[0] = *(const v8bf*)(zrh + d0);
            Ahi[kb].h[1] = *(const v8bf*)(zrh + d0 + 16);
            Alo[kb].h[0] = *(const v8bf*)(zrl + d0);
            Alo[kb].h[1] = *(const v8bf*)(zrl + d0 + 16);
        }
    }

    float minv[8];
    int   mini[8];
#pragma unroll
    for (int r = 0; r < 8; ++r) { minv[r] = 3.4e38f; mini[r] = 0; }

    const int ldC = tid >> 4;   // cooperative-load code row   (0..15)
    const int ldS = tid & 15;   // cooperative-load 16-elem segment

    // stage one 16-code hi/lo panel into LDS buffer `buf` (each thread: 64B)
    auto stage = [&](int ct, int buf) {
        const bf16* srcH = whi + (size_t)(ct * 16 + ldC) * EMB_DIM + ldS * 16;
        const bf16* srcL = wlo + (size_t)(ct * 16 + ldC) * EMB_DIM + ldS * 16;
        bf16* dH = &sWhi[buf][ldC * LDS_ROW + ldS * 16];
        bf16* dL = &sWlo[buf][ldC * LDS_ROW + ldS * 16];
#if USE_ASYNC_LDS
        GASYNC_B128(srcH,     dH);
        GASYNC_B128(srcH + 8, dH + 8);
        GASYNC_B128(srcL,     dL);
        GASYNC_B128(srcL + 8, dL + 8);
#else
        const v8bf h0 = ((const v8bf*)srcH)[0], h1 = ((const v8bf*)srcH)[1];
        const v8bf l0 = ((const v8bf*)srcL)[0], l1 = ((const v8bf*)srcL)[1];
        ((v8bf*)dH)[0] = h0; ((v8bf*)dH)[1] = h1;
        ((v8bf*)dL)[0] = l0; ((v8bf*)dL)[1] = l1;
#endif
    };

    constexpr int NT = NUM_EMB / 16;
    stage(0, 0);

    for (int ct = 0; ct < NT; ++ct) {
        const int cur = ct & 1;

        // prefetch next panel into the other buffer (overlaps this tile's WMMAs)
        if (ct + 1 < NT) stage(ct + 1, cur ^ 1);
#if USE_ASYNC_LDS
        // in-order ASYNCcnt: <=4 outstanding means tile `ct`'s 4 ops landed
        if (ct + 1 < NT) __builtin_amdgcn_s_wait_asynccnt(4);
        else             __builtin_amdgcn_s_wait_asynccnt(0);
#endif
        __syncthreads();

        const bf16* sH = sWhi[cur];
        const bf16* sL = sWlo[cur];

        // ---- 8 k-blocks x 3 WMMAs (3 independent accumulator chains)
        v8f acc0 = {}, acc1 = {}, acc2 = {};
#pragma unroll
        for (int kb = 0; kb < 8; ++kb) {
            BFrag Bhi, Blo;
            // B 32x16 bf16 layout: lane n holds K = {hh*8..hh*8+7, 16+hh*8..16+hh*8+7} of code n
            const int boff = n * LDS_ROW + kb * 32 + hh * 8;
            Bhi.h[0] = *(const v8bf*)(sH + boff);
            Bhi.h[1] = *(const v8bf*)(sH + boff + 16);
            Blo.h[0] = *(const v8bf*)(sL + boff);
            Blo.h[1] = *(const v8bf*)(sL + boff + 16);

            acc0 = __builtin_amdgcn_wmma_f32_16x16x32_bf16(
                false, Ahi[kb].v, false, Bhi.v, (short)0, acc0, false, false);
            acc1 = __builtin_amdgcn_wmma_f32_16x16x32_bf16(
                false, Ahi[kb].v, false, Blo.v, (short)0, acc1, false, false);
            acc2 = __builtin_amdgcn_wmma_f32_16x16x32_bf16(
                false, Alo[kb].v, false, Bhi.v, (short)0, acc2, false, false);
        }

        // ---- score = ||w||^2 - 2*dot ; update running per-lane argmin
        const int   cbase = ct * 16;
        const float wq    = wsq[cbase + n];
        const int   code  = cbase + n;
#pragma unroll
        for (int r = 0; r < 8; ++r) {
            const float dot   = acc0[r] + acc1[r] + acc2[r];
            const float score = wq - 2.0f * dot;
            if (score < minv[r]) { minv[r] = score; mini[r] = code; }
        }
        __syncthreads();
    }

    // ---- reduce over the 16 codes held across lanes of each half-wave
#pragma unroll
    for (int r = 0; r < 8; ++r) {
#pragma unroll
        for (int m = 8; m >= 1; m >>= 1) {
            const float ov = __shfl_xor(minv[r], m, 32);
            const int   oi = __shfl_xor(mini[r], m, 32);
            if (ov < minv[r] || (ov == minv[r] && oi < mini[r])) {
                minv[r] = ov; mini[r] = oi;
            }
        }
    }
    // C-layout: VGPR r <-> token row r (lanes 0-15) / r+8 (lanes 16-31)
    if (n == 0) {
#pragma unroll
        for (int r = 0; r < 8; ++r)
            ids[tokBase + hh * 8 + r] = mini[r];
    }
}

// ---------------------------------------------------------------------------
// Pass 5: exact fp32 gather back to [B,D,H,W]
// ---------------------------------------------------------------------------
__global__ __launch_bounds__(256) void gather_kernel(const float* __restrict__ emb,
                                                     const int* __restrict__ ids,
                                                     float* __restrict__ out) {
    const size_t i  = (size_t)blockIdx.x * 256 + threadIdx.x;
    const size_t b  = i >> 18;
    const size_t d  = (i >> 10) & 255;
    const size_t hw = i & 1023;
    const size_t t  = b * 1024 + hw;
    out[i] = emb[(size_t)ids[t] * EMB_DIM + d];
}

// ---------------------------------------------------------------------------
extern "C" void kernel_launch(void* const* d_in, const int* in_sizes, int n_in,
                              void* d_out, int out_size, void* d_ws, size_t ws_size,
                              hipStream_t stream) {
    const float* z   = (const float*)d_in[0];   // [16,256,32,32]
    const float* emb = (const float*)d_in[1];   // [8192,256]
    float*       out = (float*)d_out;

    // workspace carve-up (~25.3 MB)
    char* ws = (char*)d_ws;
    bf16*  zhi = (bf16*)(ws);                                   // 8 MB
    bf16*  zlo = (bf16*)(ws + (size_t)NELEM * 2);               // 8 MB
    bf16*  whi = (bf16*)(ws + (size_t)NELEM * 4);               // 4 MB
    bf16*  wlo = (bf16*)(ws + (size_t)NELEM * 4 + (size_t)NUM_EMB * EMB_DIM * 2);
    float* wsq = (float*)(ws + (size_t)NELEM * 4 + (size_t)NUM_EMB * EMB_DIM * 4);
    int*   ids = (int*)((char*)wsq + (size_t)NUM_EMB * 4);

    pack_z_kernel<<<NELEM / 256, 256, 0, stream>>>(z, zhi, zlo);
    pack_w_kernel<<<(NUM_EMB * EMB_DIM) / 256, 256, 0, stream>>>(emb, whi, wlo);
    wsq_kernel<<<NUM_EMB / 8, 256, 0, stream>>>(emb, wsq);
    vq_argmin_kernel<<<NTOK / 128, 256, 0, stream>>>(zhi, zlo, whi, wlo, wsq, ids);
    gather_kernel<<<NELEM / 256, 256, 0, stream>>>(emb, ids, out);
}
